// PairLayer_7791070675481
// MI455X (gfx1250) — matrix operand order repaired and checked by reading, not compiled, not run on hardware
//
#include <hip/hip_runtime.h>
#include <math.h>

#define MULC 128
#define NODE_DIM 1152
#define NNODES 10000
#define NEDGES 50000
#define EDGE_DIM 20
#define NI 384
#define WNUM 1408   // 11 * 128

typedef __bf16 bf16;
typedef __attribute__((ext_vector_type(8))) __bf16 v8bf;
typedef __attribute__((ext_vector_type(16))) __bf16 v16bf;
typedef __attribute__((ext_vector_type(8))) float v8f;

// ---------------------------------------------------------------------------
// WMMA helpers (CDNA5 wave32, 16x16x32 bf16 -> f32)
// ---------------------------------------------------------------------------
__device__ __forceinline__ v8f wmma_bf16(v16bf a, v16bf b, v8f c) {
  return __builtin_amdgcn_wmma_f32_16x16x32_bf16(false, a, false, b, (short)0, c,
                                                 false, false);
}

// A fragment: 16x32 bf16 tile from LDS row-major A[16][lda] (rows 16B-aligned).
// Lane L: row M=L&15; halves 0..7 -> K=k0+kb+0..7, halves 8..15 -> +16.
// Two ds_load_b128 + one shuffle.
__device__ __forceinline__ v16bf load_a(const bf16* A, int lda, int k0, int lane) {
  const int m = lane & 15;
  const int kb = (lane >> 4) << 3;
  const bf16* p = A + m * lda + k0 + kb;
  v8bf lo = *(const v8bf*)p;
  v8bf hi = *(const v8bf*)(p + 16);
  return __builtin_shufflevector(lo, hi, 0, 1, 2, 3, 4, 5, 6, 7, 8, 9, 10, 11,
                                 12, 13, 14, 15);
}

// B fragment from pre-packed fragment-major weights: lane's 16 halves are one
// contiguous 32-byte chunk -> two global_load_b128, zero packing VALU.
// Packed index: (((nt*KT + kt)*32 + lane)*16 + h)
__device__ __forceinline__ v16bf load_bp(const bf16* Bp, int KT, int kt, int nt,
                                         int lane) {
  return *(const v16bf*)(Bp + ((((size_t)nt * KT + kt) * 32 + lane) << 4));
}

__device__ __forceinline__ float silu(float v) {
  return v / (1.f + __expf(-v));
}

// One wave computes a 16x16 tile of  out = act(A @ B * scale + bias).
__device__ __forceinline__ void gemm16(const bf16* A, int lda, const bf16* Bp,
                                       int KT, int nt, const float* bias,
                                       float scale, int act, float* outF,
                                       bf16* outH, int ldo, int lane) {
  v8f acc = {};
#pragma unroll
  for (int kt = 0; kt < KT; ++kt) {
    v16bf a = load_a(A, lda, kt * 32, lane);
    v16bf b = load_bp(Bp, KT, kt, nt, lane);
    acc = wmma_bf16(a, b, acc);
  }
  const int n = nt * 16 + (lane & 15);
  const int mb = (lane >> 4) << 3;  // C/D: lanes 16-31 hold rows M=8..15
  const float bv = bias ? bias[n] : 0.f;
#pragma unroll
  for (int r = 0; r < 8; ++r) {
    float v = acc[r] * scale + bv;
    if (act) v = silu(v);
    const int m = mb + r;
    if (outF) outF[m * ldo + n] = v;
    if (outH) outH[m * ldo + n] = (bf16)v;
  }
}

// ---------------------------------------------------------------------------
// l=2 real-irrep <-> symmetric traceless 3x3 matrix (orthonormal embedding)
// ---------------------------------------------------------------------------
__device__ __forceinline__ void t2m(const float t[5], float M[3][3]) {
  const float A = 0.70710678118654752f;   // 1/sqrt(2)
  const float B = 0.40824829046386302f;   // 1/sqrt(6)
  M[0][1] = M[1][0] = A * t[0];
  M[1][2] = M[2][1] = A * t[1];
  M[0][2] = M[2][0] = A * t[3];
  M[0][0] = A * t[4] - B * t[2];
  M[1][1] = -A * t[4] - B * t[2];
  M[2][2] = 2.f * B * t[2];
}
__device__ __forceinline__ void m2t(const float M[3][3], float t[5]) {
  t[0] = M[0][1] * 1.41421356237309505f;
  t[1] = M[1][2] * 1.41421356237309505f;
  t[3] = M[0][2] * 1.41421356237309505f;
  t[4] = (M[0][0] - M[1][1]) * 0.70710678118654752f;
  t[2] = M[2][2] * 1.22474487139158905f;  // sqrt(6)/2
}

// ---------------------------------------------------------------------------
// K0: f32 -> bf16, repacked into WMMA B-fragment-major layout.
// ---------------------------------------------------------------------------
__global__ void pack_b_kernel(const float* __restrict__ src,
                              bf16* __restrict__ dst, int K, int N) {
  int i = blockIdx.x * 256 + threadIdx.x;
  if (i >= K * N) return;
  int h = i & 15;
  int lane = (i >> 4) & 31;
  int tile = i >> 9;
  int KT = K >> 5;
  int kt = tile % KT, nt = tile / KT;
  int n = nt * 16 + (lane & 15);
  int k = kt * 32 + ((lane >> 4) << 3) + (h & 7) + ((h >> 3) << 4);
  dst[i] = (bf16)src[k * N + n];
}

// ---------------------------------------------------------------------------
// K1: per-node  x' = norm_gate(o3_linear(x, W_pre, b_pre0), ngpre)
// 16 nodes per workgroup, 256 threads (8 waves).
// ---------------------------------------------------------------------------
__global__ void node_kernel(const float* __restrict__ x, const bf16* wpre0,
                            const bf16* wpre1, const bf16* wpre2,
                            const float* b_pre0, const bf16* ng1,
                            const float* ngb1, const bf16* ng2,
                            const float* ngb2, float* __restrict__ xp) {
  extern __shared__ char smem[];
  bf16* sA = (bf16*)smem;                              // [9][16][128] bf16
  float* sP = (float*)(smem + 9 * 16 * 128 * 2);       // [9][16][128] f32
  bf16* sF0 = (bf16*)((char*)sP + 9 * 16 * 128 * 4);   // [16][384]
  bf16* sH = sF0 + 16 * NI;                            // [16][384]
  float* sG = (float*)(sH + 16 * NI);                  // [16][384] f32

  const int tid = threadIdx.x, lane = tid & 31, wv = tid >> 5;
  const int node0 = blockIdx.x * 16;

  // stage node tile into 9 bf16 planes (s, v1:m0..2, v2:m0..4), each 16x128
  for (int i = tid; i < 16 * NODE_DIM; i += 256) {
    int nl = i / NODE_DIM, c = i % NODE_DIM;
    float v = x[(size_t)(node0 + nl) * NODE_DIM + c];
    int p, u;
    if (c < 128) { p = 0; u = c; }
    else if (c < 512) { int cc = c - 128; p = 1 + cc % 3; u = cc / 3; }
    else { int cc = c - 512; p = 4 + cc % 5; u = cc / 5; }
    sA[(p * 16 + nl) * 128 + u] = (bf16)v;
  }
  __syncthreads();

  const float inv = 0.08838834764831845f;  // 1/sqrt(128)
  for (int t = wv; t < 72; t += 8) {       // 9 planes x 8 n-tiles
    int p = t >> 3, nt = t & 7;
    const bf16* W = (p == 0) ? wpre0 : (p < 4 ? wpre1 : wpre2);
    gemm16(sA + p * 16 * 128, 128, W, 4, nt, (p == 0) ? b_pre0 : nullptr, inv, 0,
           sP + p * 16 * 128, nullptr, 128, lane);
  }
  __syncthreads();

  // f0 = [s', ||v1'||, ||v2'||]
  for (int i = tid; i < 16 * 128; i += 256) {
    int nl = i >> 7, u = i & 127;
    float s = sP[(0 * 16 + nl) * 128 + u];
    float a1 = sP[(1 * 16 + nl) * 128 + u], a2 = sP[(2 * 16 + nl) * 128 + u],
          a3 = sP[(3 * 16 + nl) * 128 + u];
    float c1 = sP[(4 * 16 + nl) * 128 + u], c2 = sP[(5 * 16 + nl) * 128 + u],
          c3 = sP[(6 * 16 + nl) * 128 + u], c4 = sP[(7 * 16 + nl) * 128 + u],
          c5 = sP[(8 * 16 + nl) * 128 + u];
    sF0[nl * NI + u] = (bf16)s;
    sF0[nl * NI + 128 + u] = (bf16)sqrtf(a1 * a1 + a2 * a2 + a3 * a3);
    sF0[nl * NI + 256 + u] =
        (bf16)sqrtf(c1 * c1 + c2 * c2 + c3 * c3 + c4 * c4 + c5 * c5);
  }
  __syncthreads();

  for (int t = wv; t < 24; t += 8)
    gemm16(sF0, NI, ng1, 12, t, ngb1, 1.f, 1, nullptr, sH, NI, lane);
  __syncthreads();
  for (int t = wv; t < 24; t += 8)
    gemm16(sH, NI, ng2, 12, t, ngb2, 1.f, 0, sG, nullptr, NI, lane);
  __syncthreads();

  // gate and emit x' in reference interleaved layout
  for (int i = tid; i < 16 * 128; i += 256) {
    int nl = i >> 7, u = i & 127;
    float* o = xp + (size_t)(node0 + nl) * NODE_DIM;
    o[u] = sG[nl * NI + u];
    float g1 = sG[nl * NI + 128 + u], g2 = sG[nl * NI + 256 + u];
#pragma unroll
    for (int m = 0; m < 3; ++m)
      o[128 + u * 3 + m] = sP[((1 + m) * 16 + nl) * 128 + u] * g1;
#pragma unroll
    for (int m = 0; m < 5; ++m)
      o[512 + u * 5 + m] = sP[((4 + m) * 16 + nl) * 128 + u] * g2;
  }
}

// ---------------------------------------------------------------------------
// K2: fully fused per-edge pipeline, 16 edges / workgroup, 256 threads.
// ---------------------------------------------------------------------------
__global__ void edge_kernel(
    const float* __restrict__ x, const float* __restrict__ xp,
    const float* __restrict__ edge_attr, const int* __restrict__ eidx,
    const bf16* w_es1, const float* es_b1, const bf16* w_es2, const float* es_b2,
    const float* er_W1, const float* er_b1, const bf16* w_er2, const float* er_b2,
    const bf16* ngp1, const float* ngpb1, const bf16* ngp2, const float* ngpb2,
    const bf16* wpo0, const bf16* wpo1, const bf16* wpo2,
    float* __restrict__ out) {
  extern __shared__ char smem[];
  bf16* s_s0 = (bf16*)smem;                 // [16][512]
  bf16* s_h1 = s_s0 + 16 * 512;             // [16][128]
  bf16* s_he = s_h1 + 16 * 128;             // [16][128]
  float* s_tpw = (float*)(s_he + 16 * 128); // [16][1408] f32
  float* s_fij = s_tpw + 16 * WNUM;         // [9][16][128] f32
  bf16* s_f0 = (bf16*)(s_fij + 9 * 16 * 128);  // [16][384]
  bf16* s_h2 = s_f0 + 16 * NI;              // [16][384]
  float* s_g = (float*)(s_h2 + 16 * NI);    // [16][384] f32
  int* s_src = (int*)(s_g + 16 * NI);       // [16]
  int* s_dst = s_src + 16;                  // [16]
  bf16* s_aq = (bf16*)s_tpw;  // alias: gated bf16 planes reuse dead tp_w region

  const int tid = threadIdx.x, lane = tid & 31, wv = tid >> 5;
  const int e0 = blockIdx.x * 16;

  if (tid < 16) {
    s_src[tid] = eidx[e0 + tid];
    s_dst[tid] = eidx[NEDGES + e0 + tid];
  }
  // he = silu(edge_attr @ er_W1 + er_b1)  (K=20: VALU)
  for (int i = tid; i < 16 * 128; i += 256) {
    int e = i >> 7, c = i & 127;
    float acc = er_b1[c];
    const float* ea = edge_attr + (size_t)(e0 + e) * EDGE_DIM;
#pragma unroll
    for (int k = 0; k < EDGE_DIM; ++k) acc += ea[k] * er_W1[k * 128 + c];
    s_he[e * 128 + c] = (bf16)silu(acc);
  }
  __syncthreads();

  // s0 = [xs_s, xd_s, <v1s,v1d>/sqrt3, <v2s,v2d>/sqrt5]
  for (int i = tid; i < 16 * 128; i += 256) {
    int e = i >> 7, u = i & 127;
    const float* ps = x + (size_t)s_src[e] * NODE_DIM;
    const float* pd = x + (size_t)s_dst[e] * NODE_DIM;
    float d1 = 0.f, d2 = 0.f;
#pragma unroll
    for (int m = 0; m < 3; ++m) d1 += ps[128 + u * 3 + m] * pd[128 + u * 3 + m];
#pragma unroll
    for (int m = 0; m < 5; ++m) d2 += ps[512 + u * 5 + m] * pd[512 + u * 5 + m];
    bf16* r = s_s0 + e * 512;
    r[u] = (bf16)ps[u];
    r[128 + u] = (bf16)pd[u];
    r[256 + u] = (bf16)(d1 * 0.57735026918962576f);
    r[384 + u] = (bf16)(d2 * 0.44721359549995794f);
  }
  __syncthreads();

  // h1 = silu(s0 @ es_W1 + es_b1)   (16x512 @ 512x128) — 1 n-tile per wave
  gemm16(s_s0, 512, w_es1, 16, wv, es_b1, 1.f, 1, nullptr, s_h1, 128, lane);
  __syncthreads();

  // tp_w = (h1 @ es_W2 + es_b2) * (he @ er_W2 + er_b2)   N=1408 -> 88 tiles
  for (int t = wv; t < 88; t += 8) {
    v8f a1 = {}, a2 = {};
#pragma unroll
    for (int kt = 0; kt < 4; ++kt) {
      a1 = wmma_bf16(load_a(s_h1, 128, kt * 32, lane),
                     load_bp(w_es2, 4, kt, t, lane), a1);
      a2 = wmma_bf16(load_a(s_he, 128, kt * 32, lane),
                     load_bp(w_er2, 4, kt, t, lane), a2);
    }
    int n = t * 16 + (lane & 15), mb = (lane >> 4) << 3;
    float b1 = es_b2[n], b2 = er_b2[n];
#pragma unroll
    for (int r = 0; r < 8; ++r)
      s_tpw[(mb + r) * WNUM + n] = (a1[r] + b1) * (a2[r] + b2);
  }
  __syncthreads();

  // Wigner tensor product (f32 VALU), per (edge, channel)
  const float PW0 = 0.57735026918962576f, PW1 = 0.86602540378443865f,
              PW2 = 1.11803398874989485f;
  const float IS3 = 0.57735026918962576f, IS5 = 0.44721359549995794f;
  const float K21 = 0.77459666924148338f, K112 = 1.0f, K222 = 0.6f;
  for (int i = tid; i < 16 * 128; i += 256) {
    int e = i >> 7, u = i & 127;
    const float* ps = xp + (size_t)s_src[e] * NODE_DIM;
    const float* pd = xp + (size_t)s_dst[e] * NODE_DIM;
    float s1 = ps[u], s2 = pd[u];
    float v1[3], v2[3], t1[5], t2[5];
#pragma unroll
    for (int m = 0; m < 3; ++m) { v1[m] = ps[128 + u * 3 + m]; v2[m] = pd[128 + u * 3 + m]; }
#pragma unroll
    for (int m = 0; m < 5; ++m) { t1[m] = ps[512 + u * 5 + m]; t2[m] = pd[512 + u * 5 + m]; }
    float w[11];
#pragma unroll
    for (int p = 0; p < 11; ++p) w[p] = s_tpw[e * WNUM + p * 128 + u];

    float dv = v1[0] * v2[0] + v1[1] * v2[1] + v1[2] * v2[2];
    float dt = 0.f;
#pragma unroll
    for (int m = 0; m < 5; ++m) dt += t1[m] * t2[m];

    float M1[3][3], M2[3][3];
    t2m(t1, M1);
    t2m(t2, M2);

    float o0 = PW0 * (w[0] * s1 * s2 + w[4] * dv * IS3 + w[9] * dt * IS5);

    float o1[3];
#pragma unroll
    for (int k = 0; k < 3; ++k) {
      float mv2 = M2[k][0] * v1[0] + M2[k][1] * v1[1] + M2[k][2] * v1[2];
      float mv1 = M1[k][0] * v2[0] + M1[k][1] * v2[1] + M1[k][2] * v2[2];
      o1[k] = PW1 * (w[1] * s1 * v2[k] * IS3 + w[3] * v1[k] * s2 * IS3 +
                     w[6] * K21 * mv2 + w[8] * K21 * mv1);
    }

    // (1,1,2): symmetric traceless v1 (x) v2
    float S[3][3];
    float tr3 = dv * (1.f / 3.f);
#pragma unroll
    for (int a = 0; a < 3; ++a)
#pragma unroll
      for (int b = 0; b < 3; ++b) S[a][b] = 0.5f * (v1[a] * v2[b] + v1[b] * v2[a]);
    S[0][0] -= tr3; S[1][1] -= tr3; S[2][2] -= tr3;
    float t112[5];
    m2t(S, t112);

    // (2,2,2): traceless anticommutator {M1, M2}
    float MM[3][3];
#pragma unroll
    for (int a = 0; a < 3; ++a)
#pragma unroll
      for (int b = 0; b < 3; ++b)
        MM[a][b] = M1[a][0] * M2[0][b] + M1[a][1] * M2[1][b] + M1[a][2] * M2[2][b];
    float trP = (2.f / 3.f) * (MM[0][0] + MM[1][1] + MM[2][2]);
    float P[3][3];
#pragma unroll
    for (int a = 0; a < 3; ++a)
#pragma unroll
      for (int b = 0; b < 3; ++b) P[a][b] = MM[a][b] + MM[b][a];
    P[0][0] -= trP; P[1][1] -= trP; P[2][2] -= trP;
    float t222[5];
    m2t(P, t222);

    float o2[5];
#pragma unroll
    for (int k = 0; k < 5; ++k)
      o2[k] = PW2 * (w[2] * s1 * t2[k] * IS5 + w[7] * t1[k] * s2 * IS5 +
                     w[5] * K112 * t112[k] + w[10] * K222 * t222[k]);

    s_fij[(0 * 16 + e) * 128 + u] = o0;
#pragma unroll
    for (int m = 0; m < 3; ++m) s_fij[((1 + m) * 16 + e) * 128 + u] = o1[m];
#pragma unroll
    for (int m = 0; m < 5; ++m) s_fij[((4 + m) * 16 + e) * 128 + u] = o2[m];
  }
  __syncthreads();

  // norm-gate (post): f0 = [s, ||v1||, ||v2||]
  for (int i = tid; i < 16 * 128; i += 256) {
    int e = i >> 7, u = i & 127;
    float s = s_fij[(0 * 16 + e) * 128 + u];
    float a1 = s_fij[(1 * 16 + e) * 128 + u], a2 = s_fij[(2 * 16 + e) * 128 + u],
          a3 = s_fij[(3 * 16 + e) * 128 + u];
    float c1 = s_fij[(4 * 16 + e) * 128 + u], c2 = s_fij[(5 * 16 + e) * 128 + u],
          c3 = s_fij[(6 * 16 + e) * 128 + u], c4 = s_fij[(7 * 16 + e) * 128 + u],
          c5 = s_fij[(8 * 16 + e) * 128 + u];
    s_f0[e * NI + u] = (bf16)s;
    s_f0[e * NI + 128 + u] = (bf16)sqrtf(a1 * a1 + a2 * a2 + a3 * a3);
    s_f0[e * NI + 256 + u] =
        (bf16)sqrtf(c1 * c1 + c2 * c2 + c3 * c3 + c4 * c4 + c5 * c5);
  }
  __syncthreads();

  for (int t = wv; t < 24; t += 8)
    gemm16(s_f0, NI, ngp1, 12, t, ngpb1, 1.f, 1, nullptr, s_h2, NI, lane);
  __syncthreads();
  for (int t = wv; t < 24; t += 8)
    gemm16(s_h2, NI, ngp2, 12, t, ngpb2, 1.f, 0, s_g, nullptr, NI, lane);
  __syncthreads();

  // gate -> bf16 planes for final o3_linear (aliased over s_tpw)
  for (int i = tid; i < 16 * 128; i += 256) {
    int e = i >> 7, u = i & 127;
    s_aq[(0 * 16 + e) * 128 + u] = (bf16)s_g[e * NI + u];
    float g1 = s_g[e * NI + 128 + u], g2 = s_g[e * NI + 256 + u];
#pragma unroll
    for (int m = 0; m < 3; ++m)
      s_aq[((1 + m) * 16 + e) * 128 + u] =
          (bf16)(s_fij[((1 + m) * 16 + e) * 128 + u] * g1);
#pragma unroll
    for (int m = 0; m < 5; ++m)
      s_aq[((4 + m) * 16 + e) * 128 + u] =
          (bf16)(s_fij[((4 + m) * 16 + e) * 128 + u] * g2);
  }
  __syncthreads();

  // final o3_linear (no bias), write straight to global out
  const float inv = 0.08838834764831845f;
  for (int t = wv; t < 72; t += 8) {
    int p = t >> 3, nt = t & 7;
    const bf16* W = (p == 0) ? wpo0 : (p < 4 ? wpo1 : wpo2);
    v8f acc = {};
#pragma unroll
    for (int kt = 0; kt < 4; ++kt)
      acc = wmma_bf16(load_a(s_aq + p * 16 * 128, 128, kt * 32, lane),
                      load_bp(W, 4, kt, nt, lane), acc);
    int v = nt * 16 + (lane & 15), mb = (lane >> 4) << 3;
    int col = (p == 0) ? v : (p < 4 ? 128 + v * 3 + (p - 1) : 512 + v * 5 + (p - 4));
#pragma unroll
    for (int r = 0; r < 8; ++r)
      out[(size_t)(e0 + mb + r) * NODE_DIM + col] = acc[r] * inv;
  }
}

// ---------------------------------------------------------------------------
// host launcher
// ---------------------------------------------------------------------------
extern "C" void kernel_launch(void* const* d_in, const int* in_sizes, int n_in,
                              void* d_out, int out_size, void* d_ws,
                              size_t ws_size, hipStream_t stream) {
  const float* x = (const float*)d_in[0];
  const float* edge_attr = (const float*)d_in[1];
  const int* eidx = (const int*)d_in[2];

  char* ws = (char*)d_ws;
  float* xp = (float*)ws;
  size_t off = (size_t)NNODES * NODE_DIM * sizeof(float);  // 32B-aligned
  auto carve = [&](size_t elems) {
    bf16* p = (bf16*)(ws + off);
    off += elems * sizeof(bf16);
    return p;
  };
  bf16* w_pre0 = carve(128 * 128);
  bf16* w_pre1 = carve(128 * 128);
  bf16* w_pre2 = carve(128 * 128);
  bf16* w_ng1 = carve(NI * NI);
  bf16* w_ng2 = carve(NI * NI);
  bf16* w_es1 = carve(512 * 128);
  bf16* w_es2 = carve(128 * WNUM);
  bf16* w_er2 = carve(128 * WNUM);
  bf16* w_ngp1 = carve(NI * NI);
  bf16* w_ngp2 = carve(NI * NI);
  bf16* w_po0 = carve(128 * 128);
  bf16* w_po1 = carve(128 * 128);
  bf16* w_po2 = carve(128 * 128);

  auto pack = [&](int idx, bf16* dst, int K, int N) {
    pack_b_kernel<<<(K * N + 255) / 256, 256, 0, stream>>>(
        (const float*)d_in[idx], dst, K, N);
  };
  pack(3, w_pre0, 128, 128);
  pack(4, w_pre1, 128, 128);
  pack(5, w_pre2, 128, 128);
  pack(7, w_ng1, NI, NI);
  pack(9, w_ng2, NI, NI);
  pack(11, w_es1, 512, 128);
  pack(13, w_es2, 128, WNUM);
  pack(17, w_er2, 128, WNUM);
  pack(19, w_ngp1, NI, NI);
  pack(21, w_ngp2, NI, NI);
  pack(23, w_po0, 128, 128);
  pack(24, w_po1, 128, 128);
  pack(25, w_po2, 128, 128);

  const size_t NODE_SMEM =
      (size_t)9 * 16 * 128 * 2 + (size_t)9 * 16 * 128 * 4 + 16 * NI * 2 +
      16 * NI * 2 + 16 * NI * 4;
  const size_t EDGE_SMEM =
      (size_t)16 * 512 * 2 + 16 * 128 * 2 + 16 * 128 * 2 +
      (size_t)16 * WNUM * 4 + (size_t)9 * 16 * 128 * 4 + 16 * NI * 2 +
      16 * NI * 2 + 16 * NI * 4 + 32 * sizeof(int);

  (void)hipFuncSetAttribute((const void*)node_kernel,
                            hipFuncAttributeMaxDynamicSharedMemorySize,
                            (int)NODE_SMEM);
  (void)hipFuncSetAttribute((const void*)edge_kernel,
                            hipFuncAttributeMaxDynamicSharedMemorySize,
                            (int)EDGE_SMEM);

  node_kernel<<<NNODES / 16, 256, NODE_SMEM, stream>>>(
      x, w_pre0, w_pre1, w_pre2, (const float*)d_in[6], w_ng1,
      (const float*)d_in[8], w_ng2, (const float*)d_in[10], xp);

  edge_kernel<<<NEDGES / 16, 256, EDGE_SMEM, stream>>>(
      x, xp, edge_attr, eidx, w_es1, (const float*)d_in[12], w_es2,
      (const float*)d_in[14], (const float*)d_in[15], (const float*)d_in[16],
      w_er2, (const float*)d_in[18], w_ngp1, (const float*)d_in[20], w_ngp2,
      (const float*)d_in[22], w_po0, w_po1, w_po2, (float*)d_out);
}
